// GNNEncoder_68650757259334
// MI455X (gfx1250) — compile-verified, matrix operand order checked by reading, not run on hardware
//
#include <hip/hip_runtime.h>
#include <hip/hip_bf16.h>
#include <math.h>

// Problem constants (match reference)
#define F_IN   64
#define H_DIM  256
#define S_DIM  32
#define L_DIM  128
#define B_DIM  16
#define EPS_F  1e-9f

typedef __attribute__((ext_vector_type(16))) _Float16 v16h;
typedef __attribute__((ext_vector_type(8)))  _Float16 v8h;
typedef __attribute__((ext_vector_type(8)))  float    v8f;
typedef int v4i __attribute__((vector_size(16)));   // matches builtin pointee type

#define AS1 __attribute__((address_space(1)))
#define AS3 __attribute__((address_space(3)))

#if defined(__has_builtin)
#if __has_builtin(__builtin_amdgcn_global_load_async_to_lds_b128)
#define HAVE_ASYNC_LDS 1
#endif
#endif

static __device__ __forceinline__ void wait_async0() {
#if defined(HAVE_ASYNC_LDS)
  asm volatile("" ::: "memory");
#if __has_builtin(__builtin_amdgcn_s_wait_asynccnt)
  __builtin_amdgcn_s_wait_asynccnt(0);
#else
  asm volatile("s_wait_asynccnt 0" ::: "memory");
#endif
  asm volatile("" ::: "memory");
#endif
}

// copy one 16-byte chunk from global to LDS (async path when available)
static __device__ __forceinline__ void copy16_g2l(const _Float16* g, _Float16* l) {
#if defined(HAVE_ASYNC_LDS)
  __builtin_amdgcn_global_load_async_to_lds_b128(
      (AS1 v4i*)(unsigned long long)(const void*)g,
      (AS3 v4i*)(unsigned)(unsigned long long)(const void*)l,
      0, 0);
#else
  *(v8h*)l = *(const v8h*)g;
#endif
}

static __device__ __forceinline__ v16h cat8(v8h lo, v8h hi) {
  return __builtin_shufflevector(lo, hi, 0,1,2,3,4,5,6,7,8,9,10,11,12,13,14,15);
}
static __device__ __forceinline__ v8f wmma16(v16h a, v16h b, v8f c) {
  // D = A(16x32 f16) * B(32x16 f16) + C(16x16 f32)
  return __builtin_amdgcn_wmma_f32_16x16x32_f16(false, a, false, b, (short)0, c, false, false);
}
template<int NT> static __device__ __forceinline__ void zero_acc(v8f* acc) {
#pragma unroll
  for (int nt = 0; nt < NT; ++nt) {
    v8f z = {0.f,0.f,0.f,0.f,0.f,0.f,0.f,0.f};
    acc[nt] = z;
  }
}

// A is [16][K] f16 row-major in LDS (per-wave private). W is fragment-packed:
// [kt][nt][lane][16] halfs, so each lane loads its B fragment as one v16h.
template<int K, int NT>
static __device__ __forceinline__ void gemm_lds(const _Float16* aBase,
                                                const _Float16* __restrict__ wp,
                                                v8f* acc, int lane) {
  const int m    = lane & 15;
  const int koff = (lane & 16) ? 8 : 0;   // A-fragment K base per 16x32 16-bit layout
  for (int kt = 0; kt < K / 32; ++kt) {
    const _Float16* ar = aBase + (size_t)m * K + kt * 32 + koff;
    v16h af = cat8(*(const v8h*)ar, *(const v8h*)(ar + 16)); // K: base..+7, base+16..+23
#pragma unroll
    for (int nt = 0; nt < NT; ++nt) {
      v16h bf = *(const v16h*)(wp + ((size_t)(kt * NT + nt) * 32 + lane) * 16);
      acc[nt] = wmma16(af, bf, acc[nt]);
    }
  }
}

// bias + relu + store 16x256 f32 accumulators as f16 row-major into LDS.
// D layout: vgpr r -> M = r + 8*(lane>=16); N = (lane&15) + 16*nt
template<int NT>
static __device__ __forceinline__ void bias_relu_store_lds(v8f* acc,
                                                           const float* __restrict__ bias,
                                                           _Float16* hOut, int lane) {
  const int mD = (lane & 16) ? 8 : 0;
  const int nB = lane & 15;
#pragma unroll
  for (int nt = 0; nt < NT; ++nt) {
    float bn = bias[nt * 16 + nB];
#pragma unroll
    for (int r = 0; r < 8; ++r) {
      float v = fmaxf(acc[nt][r] + bn, 0.f);
      hOut[(size_t)(r + mD) * H_DIM + nt * 16 + nB] = (_Float16)v;
    }
  }
}

// ---------------------------------------------------------------------------
// Edge message MLP: per 16-edge tile (one wave): gather concat(x[dst],x[src])
// into LDS (async-to-LDS when available), 3-stage WMMA MLP
// (KCAT->256 relu, 256->256 relu, 256->256), atomic scatter-add into out[dst].
// ---------------------------------------------------------------------------
template<int KCAT, int WAVES>
__global__ __launch_bounds__(WAVES * 32)
void edge_mlp_kernel(const _Float16* __restrict__ xh,
                     const int* __restrict__ srcI, const int* __restrict__ dstI,
                     const _Float16* __restrict__ w1p, const float* __restrict__ b1,
                     const _Float16* __restrict__ w2p, const float* __restrict__ b2,
                     const _Float16* __restrict__ w3p, const float* __restrict__ b3,
                     float* __restrict__ outAcc, int numEdges) {
  __shared__ alignas(32) _Float16 sX[WAVES][16 * KCAT];
  __shared__ alignas(32) _Float16 sH[WAVES][16 * H_DIM];
  __shared__ int sDst[WAVES][16];
  const int lane = threadIdx.x & 31;
  const int wave = threadIdx.x >> 5;
  const int tile = blockIdx.x * WAVES + wave;
  const int eBase = tile * 16;
  if (eBase >= numEdges) return;

  constexpr int KIN    = KCAT / 2;   // per-node feature width
  constexpr int CHUNKS = KCAT / 8;   // 8-half chunks per concat row
  constexpr int CPN    = KIN / 8;    // chunks coming from dst half

  // cooperative gather, row-uniform edge indices (scalar loads)
#pragma unroll
  for (int m = 0; m < 16; ++m) {
    const int e = eBase + m;
    const int dn = dstI[e];
    const int sn = srcI[e];
    for (int c = lane; c < CHUNKS; c += 32) {
      const int node = (c < CPN) ? dn : sn;
      const int off  = (c < CPN) ? (c * 8) : ((c - CPN) * 8);
      copy16_g2l(xh + (size_t)node * KIN + off,
                 &sX[wave][(size_t)m * KCAT + c * 8]);
    }
  }
  if (lane < 16) sDst[wave][lane] = dstI[eBase + lane];
  wait_async0();   // async gather (if used) must land before DS reads

  v8f acc[16];
  // stage 1: concat(2*KIN) -> 256, relu
  zero_acc<16>(acc);
  gemm_lds<KCAT, 16>(&sX[wave][0], w1p, acc, lane);
  bias_relu_store_lds<16>(acc, b1, &sH[wave][0], lane);
  // stage 2: 256 -> 256, relu (sH reused in place: wave-private LDS, in-order DS)
  zero_acc<16>(acc);
  gemm_lds<H_DIM, 16>(&sH[wave][0], w2p, acc, lane);
  bias_relu_store_lds<16>(acc, b2, &sH[wave][0], lane);
  // stage 3: 256 -> 256, scatter-add message to dst node
  zero_acc<16>(acc);
  gemm_lds<H_DIM, 16>(&sH[wave][0], w3p, acc, lane);
  const int mD = (lane & 16) ? 8 : 0;
  const int nB = lane & 15;
#pragma unroll
  for (int nt = 0; nt < 16; ++nt) {
    float bn = b3[nt * 16 + nB];
#pragma unroll
    for (int r = 0; r < 8; ++r) {
      int d = sDst[wave][r + mD];
      atomicAdd(outAcc + (size_t)d * H_DIM + nt * 16 + nB, acc[nt][r] + bn);
    }
  }
}

// ---------------------------------------------------------------------------
// Dense 2-stage WMMA MLP: rows x 256 -> relu(256) -> NOUT (=16*NT2), f32 out.
// Stage-1 A fragments load straight from global f16 rows.
// ---------------------------------------------------------------------------
template<int NT2, int WAVES>
__global__ __launch_bounds__(WAVES * 32)
void dense_mlp_kernel(const _Float16* __restrict__ A,
                      const _Float16* __restrict__ w1p, const float* __restrict__ b1,
                      const _Float16* __restrict__ w2p, const float* __restrict__ b2,
                      float* __restrict__ outB, int rows) {
  __shared__ alignas(32) _Float16 sH[WAVES][16 * H_DIM];
  const int lane = threadIdx.x & 31;
  const int wave = threadIdx.x >> 5;
  const int tile = blockIdx.x * WAVES + wave;
  const int rBase = tile * 16;
  if (rBase >= rows) return;

  const int m    = lane & 15;
  const int koff = (lane & 16) ? 8 : 0;
  v8f acc[16];
  zero_acc<16>(acc);
  for (int kt = 0; kt < H_DIM / 32; ++kt) {
    const _Float16* ar = A + (size_t)(rBase + m) * H_DIM + kt * 32 + koff;
    v16h af = cat8(*(const v8h*)ar, *(const v8h*)(ar + 16));
#pragma unroll
    for (int nt = 0; nt < 16; ++nt) {
      v16h bf = *(const v16h*)(w1p + ((size_t)(kt * 16 + nt) * 32 + lane) * 16);
      acc[nt] = wmma16(af, bf, acc[nt]);
    }
  }
  bias_relu_store_lds<16>(acc, b1, &sH[wave][0], lane);

  v8f acc2[NT2];
  zero_acc<NT2>(acc2);
  gemm_lds<H_DIM, NT2>(&sH[wave][0], w2p, acc2, lane);
  const int mD = (lane & 16) ? 8 : 0;
  const int nB = lane & 15;
  constexpr int NOUT = NT2 * 16;
#pragma unroll
  for (int nt = 0; nt < NT2; ++nt) {
    float bn = b2[nt * 16 + nB];
#pragma unroll
    for (int r = 0; r < 8; ++r)
      outB[(size_t)(rBase + r + mD) * NOUT + nt * 16 + nB] = acc2[nt][r] + bn;
  }
}

// ---------------------------------------------------------------------------
// Weight packer: w[K][Nout] f32 -> fragment order [kt][nt][lane][16] f16,
// following the 16-bit 32x16 B-matrix VGPR layout.
// ---------------------------------------------------------------------------
__global__ __launch_bounds__(256)
void pack_weight_kernel(const float* __restrict__ w, _Float16* __restrict__ wp,
                        int K, int Nout) {
  int t = blockIdx.x * 256 + threadIdx.x;
  int nTiles = Nout >> 4;
  int total = (K >> 5) * nTiles * 32;
  if (t >= total) return;
  int lane = t & 31;
  int grp = t >> 5;
  int nt = grp % nTiles;
  int kt = grp / nTiles;
  int n = nt * 16 + (lane & 15);
  int kbase = kt * 32 + ((lane & 16) ? 8 : 0);
  v16h v;
#pragma unroll
  for (int h = 0; h < 16; ++h) {
    int k = kbase + (h < 8 ? h : h + 8);
    v[h] = (_Float16)w[(size_t)k * Nout + n];
  }
  *(v16h*)(wp + (size_t)t * 16) = v;
}

// relu + LayerNorm over H=256, one node per 256-thread block.
__global__ __launch_bounds__(256)
void relu_ln_kernel(const float* __restrict__ acc,
                    const float* __restrict__ g, const float* __restrict__ b,
                    float* __restrict__ xf32, _Float16* __restrict__ xh) {
  const int n = blockIdx.x;
  const int t = threadIdx.x;
  __shared__ float red[H_DIM];
  float v = fmaxf(acc[(size_t)n * H_DIM + t], 0.f);
  red[t] = v; __syncthreads();
  for (int s2 = H_DIM / 2; s2 > 0; s2 >>= 1) { if (t < s2) red[t] += red[t + s2]; __syncthreads(); }
  float mean = red[0] * (1.f / H_DIM);
  __syncthreads();
  float d = v - mean;
  red[t] = d * d; __syncthreads();
  for (int s2 = H_DIM / 2; s2 > 0; s2 >>= 1) { if (t < s2) red[t] += red[t + s2]; __syncthreads(); }
  float var = red[0] * (1.f / H_DIM);
  float y = d * rsqrtf(var + 1e-5f) * g[t] + b[t];
  if (xf32) xf32[(size_t)n * H_DIM + t] = y;
  xh[(size_t)n * H_DIM + t] = (_Float16)y;
}

// Gumbel-softmax over S=32 (one wave per node), block-accumulated loss stats.
// lossAcc[0..31] = sum_n s[n][j];  lossAcc[32] = sum_n sum_j s*log(s+eps)
__global__ __launch_bounds__(256)
void gumbel_softmax_kernel(const float* __restrict__ logits, const float* __restrict__ u,
                           float* __restrict__ sOut, float* __restrict__ lossAcc,
                           int nNodes) {
  const int lane = threadIdx.x & 31;
  const int wave = threadIdx.x >> 5;
  const int n = blockIdx.x * 8 + wave;
  __shared__ float blkAvg[32];
  __shared__ float blkEnt;
  if (threadIdx.x < 32) blkAvg[threadIdx.x] = 0.f;
  if (threadIdx.x == 0) blkEnt = 0.f;
  __syncthreads();
  float sval = 0.f, entc = 0.f;
  if (n < nNodes) {
    float gum = -logf(-logf(u[(size_t)n * S_DIM + lane] + EPS_F) + EPS_F);
    float z = logits[(size_t)n * S_DIM + lane] + gum;   // TAU = 1
    float mx = z;
    for (int o = 16; o > 0; o >>= 1) mx = fmaxf(mx, __shfl_xor(mx, o, 32));
    float p = expf(z - mx);
    float sum = p;
    for (int o = 16; o > 0; o >>= 1) sum += __shfl_xor(sum, o, 32);
    sval = p / sum;
    sOut[(size_t)n * S_DIM + lane] = sval;
    entc = sval * logf(sval + EPS_F);
  }
  atomicAdd(&blkAvg[lane], sval);
  float es = entc;
  for (int o = 16; o > 0; o >>= 1) es += __shfl_xor(es, o, 32);
  if (lane == 0) atomicAdd(&blkEnt, es);
  __syncthreads();
  if (threadIdx.x < 32) atomicAdd(&lossAcc[threadIdx.x], blkAvg[threadIdx.x]);
  if (threadIdx.x == 0) atomicAdd(&lossAcc[32], blkEnt);
}

// pooled[batch[n], j, h] += x[n,h] * s[n,j]   (one node per 256-thread block)
__global__ __launch_bounds__(256)
void pool_kernel(const float* __restrict__ x, const float* __restrict__ s,
                 const int* __restrict__ batch, float* __restrict__ pooled) {
  const int n = blockIdx.x;
  const int t = threadIdx.x;
  __shared__ float sv[S_DIM];
  if (t < S_DIM) sv[t] = s[(size_t)n * S_DIM + t];
  __syncthreads();
  float xv = x[(size_t)n * H_DIM + t];
  int bb = batch[n];
  float* base = pooled + (size_t)bb * S_DIM * H_DIM + t;
#pragma unroll
  for (int j = 0; j < S_DIM; ++j) atomicAdd(base + (size_t)j * H_DIM, xv * sv[j]);
}

__global__ void finalize_loss_kernel(const float* __restrict__ lossAcc,
                                     float* __restrict__ outLoss, int nNodes) {
  int lane = threadIdx.x;
  float avg = lossAcc[lane] / (float)nNodes;
  float dv = avg * logf(avg + EPS_F);
  for (int o = 16; o > 0; o >>= 1) dv += __shfl_xor(dv, o, 32);
  if (lane == 0) {
    float entropy = -lossAcc[32] / (float)nNodes;
    outLoss[0] = entropy + dv;
  }
}

__global__ void f32_to_f16_kernel(const float* __restrict__ in, _Float16* __restrict__ out,
                                  long long n) {
  long long i = (long long)blockIdx.x * blockDim.x + threadIdx.x;
  long long stride = (long long)gridDim.x * blockDim.x;
  for (; i < n; i += stride) out[i] = (_Float16)in[i];
}
__global__ void fill_zero_kernel(float* __restrict__ p, long long n) {
  long long i = (long long)blockIdx.x * blockDim.x + threadIdx.x;
  long long stride = (long long)gridDim.x * blockDim.x;
  for (; i < n; i += stride) p[i] = 0.f;
}

// ---------------------------------------------------------------------------
extern "C" void kernel_launch(void* const* d_in, const int* in_sizes, int n_in,
                              void* d_out, int out_size, void* d_ws, size_t ws_size,
                              hipStream_t stream) {
  const float* x     = (const float*)d_in[0];
  const float* u     = (const float*)d_in[1];
  const int*   ei    = (const int*)d_in[2];
  const int*   batch = (const int*)d_in[3];
  const float* g1w1 = (const float*)d_in[4];  const float* g1b1 = (const float*)d_in[5];
  const float* g1w2 = (const float*)d_in[6];  const float* g1b2 = (const float*)d_in[7];
  const float* g1w3 = (const float*)d_in[8];  const float* g1b3 = (const float*)d_in[9];
  const float* ln1g = (const float*)d_in[10]; const float* ln1b = (const float*)d_in[11];
  const float* g2w1 = (const float*)d_in[12]; const float* g2b1 = (const float*)d_in[13];
  const float* g2w2 = (const float*)d_in[14]; const float* g2b2 = (const float*)d_in[15];
  const float* g2w3 = (const float*)d_in[16]; const float* g2b3 = (const float*)d_in[17];
  const float* ln2g = (const float*)d_in[18]; const float* ln2b = (const float*)d_in[19];
  const float* aw1  = (const float*)d_in[20]; const float* ab1  = (const float*)d_in[21];
  const float* aw2  = (const float*)d_in[22]; const float* ab2  = (const float*)d_in[23];
  const float* ow1  = (const float*)d_in[24]; const float* ob1  = (const float*)d_in[25];
  const float* ow2  = (const float*)d_in[26]; const float* ob2  = (const float*)d_in[27];

  const int N = in_sizes[3];          // 20000
  const int E = in_sizes[2] / 2;      // 320000
  const int* srcI = ei;
  const int* dstI = ei + E;

  // workspace bump allocator (256B aligned)
  char* ws = (char*)d_ws;
  size_t off = 0;
  auto alloc = [&](size_t bytes) -> char* {
    char* p = ws + off;
    off += (bytes + 255) & ~(size_t)255;
    return p;
  };
  _Float16* xh0     = (_Float16*)alloc((size_t)N * F_IN * 2);
  _Float16* x1h     = (_Float16*)alloc((size_t)N * H_DIM * 2);
  _Float16* x2h     = (_Float16*)alloc((size_t)N * H_DIM * 2);
  float*    accum   = (float*)alloc((size_t)N * H_DIM * 4);     // layer accum, reused; x2 f32 in place
  float*    logits  = (float*)alloc((size_t)N * S_DIM * 4);
  float*    pooled  = (float*)alloc((size_t)B_DIM * S_DIM * H_DIM * 4);
  _Float16* pooledh = (_Float16*)alloc((size_t)B_DIM * S_DIM * H_DIM * 2);
  float*    lossAcc = (float*)alloc(64 * 4);
  _Float16* pg1w1 = (_Float16*)alloc((size_t)2 * F_IN * H_DIM * 2);
  _Float16* pg1w2 = (_Float16*)alloc((size_t)H_DIM * H_DIM * 2);
  _Float16* pg1w3 = (_Float16*)alloc((size_t)H_DIM * H_DIM * 2);
  _Float16* pg2w1 = (_Float16*)alloc((size_t)2 * H_DIM * H_DIM * 2);
  _Float16* pg2w2 = (_Float16*)alloc((size_t)H_DIM * H_DIM * 2);
  _Float16* pg2w3 = (_Float16*)alloc((size_t)H_DIM * H_DIM * 2);
  _Float16* paw1  = (_Float16*)alloc((size_t)H_DIM * H_DIM * 2);
  _Float16* paw2  = (_Float16*)alloc((size_t)H_DIM * S_DIM * 2);
  _Float16* pow1  = (_Float16*)alloc((size_t)H_DIM * H_DIM * 2);
  _Float16* pow2  = (_Float16*)alloc((size_t)H_DIM * L_DIM * 2);

  float* latentOut = (float*)d_out;                          // [B,S,L] = 65536
  float* sOut      = latentOut + (size_t)B_DIM * S_DIM * L_DIM;
  float* lossOut   = sOut + (size_t)N * S_DIM;

  auto pack = [&](const float* w, _Float16* wp, int K, int Nout) {
    int total = (K / 32) * (Nout / 16) * 32;
    pack_weight_kernel<<<(total + 255) / 256, 256, 0, stream>>>(w, wp, K, Nout);
  };
  pack(g1w1, pg1w1, 2 * F_IN, H_DIM);
  pack(g1w2, pg1w2, H_DIM, H_DIM);
  pack(g1w3, pg1w3, H_DIM, H_DIM);
  pack(g2w1, pg2w1, 2 * H_DIM, H_DIM);
  pack(g2w2, pg2w2, H_DIM, H_DIM);
  pack(g2w3, pg2w3, H_DIM, H_DIM);
  pack(aw1,  paw1,  H_DIM, H_DIM);
  pack(aw2,  paw2,  H_DIM, S_DIM);
  pack(ow1,  pow1,  H_DIM, H_DIM);
  pack(ow2,  pow2,  H_DIM, L_DIM);

  f32_to_f16_kernel<<<1024, 256, 0, stream>>>(x, xh0, (long long)N * F_IN);

  // ---- GNN layer 1 ----
  fill_zero_kernel<<<2048, 256, 0, stream>>>(accum, (long long)N * H_DIM);
  {
    int tiles = (E + 15) / 16;
    edge_mlp_kernel<2 * F_IN, 4><<<(tiles + 3) / 4, 4 * 32, 0, stream>>>(
        xh0, srcI, dstI, pg1w1, g1b1, pg1w2, g1b2, pg1w3, g1b3, accum, E);
  }
  relu_ln_kernel<<<N, H_DIM, 0, stream>>>(accum, ln1g, ln1b, (float*)nullptr, x1h);

  // ---- GNN layer 2 ----
  fill_zero_kernel<<<2048, 256, 0, stream>>>(accum, (long long)N * H_DIM);
  {
    int tiles = (E + 15) / 16;
    edge_mlp_kernel<2 * H_DIM, 2><<<(tiles + 1) / 2, 2 * 32, 0, stream>>>(
        x1h, srcI, dstI, pg2w1, g2b1, pg2w2, g2b2, pg2w3, g2b3, accum, E);
  }
  relu_ln_kernel<<<N, H_DIM, 0, stream>>>(accum, ln2g, ln2b, accum /*x2 f32 in place*/, x2h);

  // ---- assignment logits ----
  {
    int tiles = (N + 15) / 16;
    dense_mlp_kernel<S_DIM / 16, 4><<<(tiles + 3) / 4, 4 * 32, 0, stream>>>(
        x2h, paw1, ab1, paw2, ab2, logits, N);
  }

  // ---- Gumbel-softmax + loss stats ----
  fill_zero_kernel<<<1, 64, 0, stream>>>(lossAcc, 33);
  gumbel_softmax_kernel<<<(N + 7) / 8, 256, 0, stream>>>(logits, u, sOut, lossAcc, N);

  // ---- pooling ----
  fill_zero_kernel<<<512, 256, 0, stream>>>(pooled, (long long)B_DIM * S_DIM * H_DIM);
  pool_kernel<<<N, H_DIM, 0, stream>>>(accum /*x2*/, sOut, batch, pooled);
  f32_to_f16_kernel<<<512, 256, 0, stream>>>(pooled, pooledh, (long long)B_DIM * S_DIM * H_DIM);

  // ---- output MLP ----
  {
    int rows = B_DIM * S_DIM;  // 512
    int tiles = rows / 16;     // 32
    dense_mlp_kernel<L_DIM / 16, 4><<<(tiles + 3) / 4, 4 * 32, 0, stream>>>(
        pooledh, pow1, ob1, pow2, ob2, latentOut, rows);
  }

  finalize_loss_kernel<<<1, 32, 0, stream>>>(lossAcc, lossOut, N);
}
